// ADD_GCN_47347719471351
// MI455X (gfx1250) — compile-verified, matrix operand order checked by reading, not run on hardware
//
#include <hip/hip_runtime.h>
#include <hip/hip_bf16.h>
#include <stdint.h>

// ---------------- problem constants ----------------
#define NB    8
#define NP    128
#define NC    4
#define CIN   256
#define HW    1024
#define DOBJ  4424
#define DPAIR 8848          // 2*DOBJ
#define CTOT  8976          // 128 + 8848
#define KPAD  8992          // CTOT padded to multiple of 32
#define BP    1024          // NB*NP
#define NCOL  4096          // BP*NC

typedef __attribute__((ext_vector_type(16))) __bf16 v16bf;
typedef __attribute__((ext_vector_type(8)))  float  v8f;

static __device__ __forceinline__ float lrelu(float x){ return x > 0.f ? x : 0.2f * x; }
static __device__ __forceinline__ float sigmoidf_(float x){ return 1.f / (1.f + expf(-x)); }
static __device__ __forceinline__ __bf16 f2bf(float f){
  uint32_t u = __builtin_bit_cast(uint32_t, f);
  uint32_t r = u + 0x7FFFu + ((u >> 16) & 1u);
  uint16_t h = (uint16_t)(r >> 16);
  return __builtin_bit_cast(__bf16, h);
}
static __device__ __forceinline__ float bf2f(__bf16 b){
  uint16_t h = __builtin_bit_cast(uint16_t, b);
  uint32_t u = ((uint32_t)h) << 16;
  return __builtin_bit_cast(float, u);
}

// ---------------- stage 1: fc head: y = fc_w @ x, out1 = max_s y ----------------
__global__ __launch_bounds__(256) void k_y_out1(const float* __restrict__ x,
                                                const float* __restrict__ fcw,
                                                float* __restrict__ yws,
                                                float* __restrict__ out1)
{
  __shared__ float wsm[NC * CIN];
  __shared__ float red[256];
  const int b = blockIdx.x, tid = threadIdx.x;
  for (int i = tid; i < NC * CIN; i += 256) wsm[i] = fcw[i];
  __syncthreads();
  float mx[NC] = {-1e30f, -1e30f, -1e30f, -1e30f};
  for (int s = tid; s < HW; s += 256){
    const float* xb = x + (size_t)b * CIN * HW + s;
    float a[NC] = {0.f, 0.f, 0.f, 0.f};
    for (int c = 0; c < CIN; ++c){
      float xv = xb[(size_t)c * HW];
      #pragma unroll
      for (int o = 0; o < NC; ++o) a[o] += wsm[o * CIN + c] * xv;
    }
    #pragma unroll
    for (int o = 0; o < NC; ++o){
      yws[((size_t)(b * NC + o)) * HW + s] = a[o];
      mx[o] = fmaxf(mx[o], a[o]);
    }
  }
  for (int o = 0; o < NC; ++o){
    red[tid] = mx[o];
    __syncthreads();
    for (int st = 128; st > 0; st >>= 1){
      if (tid < st) red[tid] = fmaxf(red[tid], red[tid + st]);
      __syncthreads();
    }
    if (tid == 0) out1[b * NC + o] = red[0];
    __syncthreads();
  }
}

// ---------------- stage 2: fused SAM: v[b,c,n] = sum_s (ct_w@x + ct_b) * sigmoid(y) ----------------
// also tnode[b,c,n] = leaky(sum_m A[n,m] v[b,c,m])
__global__ __launch_bounds__(256) void k_vsam(const float* __restrict__ x,
                                              const float* __restrict__ ctw,
                                              const float* __restrict__ ctb,
                                              const float* __restrict__ yws,
                                              const float* __restrict__ A,
                                              float* __restrict__ vws,
                                              float* __restrict__ tnode)
{
  __shared__ float wsm[CIN];
  __shared__ float red[NC * 256];
  const int blk = blockIdx.x;
  const int b = blk >> 7, c = blk & 127, tid = threadIdx.x;
  for (int i = tid; i < CIN; i += 256) wsm[i] = ctw[c * CIN + i];
  __syncthreads();
  const float bias = ctb[c];
  float an[NC] = {0.f, 0.f, 0.f, 0.f};
  for (int s = tid; s < HW; s += 256){
    const float* xb = x + (size_t)b * CIN * HW + s;
    float ctxv = bias;
    for (int ci = 0; ci < CIN; ++ci) ctxv += wsm[ci] * xb[(size_t)ci * HW];
    #pragma unroll
    for (int n = 0; n < NC; ++n){
      float yv = yws[((size_t)(b * NC + n)) * HW + s];
      an[n] += ctxv * sigmoidf_(yv);
    }
  }
  #pragma unroll
  for (int n = 0; n < NC; ++n) red[n * 256 + tid] = an[n];
  __syncthreads();
  for (int st = 128; st > 0; st >>= 1){
    if (tid < st){
      #pragma unroll
      for (int n = 0; n < NC; ++n) red[n * 256 + tid] += red[n * 256 + tid + st];
    }
    __syncthreads();
  }
  if (tid == 0){
    float vv[NC];
    #pragma unroll
    for (int n = 0; n < NC; ++n){
      vv[n] = red[n * 256];
      vws[((b * NP + c) << 2) + n] = vv[n];
    }
    #pragma unroll
    for (int n = 0; n < NC; ++n){
      float t = 0.f;
      #pragma unroll
      for (int m = 0; m < NC; ++m) t += A[n * NC + m] * vv[m];
      tnode[((b * NP + c) << 2) + n] = lrelu(t);
    }
  }
}

// ---------------- stage 3: gather pair features -> bf16 ----------------
__global__ __launch_bounds__(256) void k_gather(const int* __restrict__ idx,
                                                const float* __restrict__ obj,
                                                __bf16* __restrict__ pf)
{
  const int p = blockIdx.x, tid = threadIdx.x;
  const int i0 = idx[p * 2 + 0];
  const int i1 = idx[p * 2 + 1];
  const float* r0 = obj + (size_t)i0 * DOBJ;
  const float* r1 = obj + (size_t)i1 * DOBJ;
  __bf16* dst = pf + (size_t)p * DPAIR;
  for (int c = tid; c < DOBJ; c += 256) dst[c] = f2bf(r0[c]);
  for (int c = tid; c < DOBJ; c += 256) dst[DOBJ + c] = f2bf(r1[c]);
}

// ---------------- stage 4: build X1[pn][k] = leaky(A-node-mix of g) in bf16 ----------------
__global__ __launch_bounds__(256) void k_buildX1(const __bf16* __restrict__ pf,
                                                 const float* __restrict__ tnode,
                                                 const float* __restrict__ A,
                                                 __bf16* __restrict__ X1)
{
  const int pn = blockIdx.x;
  const int p = pn >> 2, n = pn & 3, b = p >> 7, tid = threadIdx.x;
  const float rsA = A[n * NC + 0] + A[n * NC + 1] + A[n * NC + 2] + A[n * NC + 3];
  __bf16* dst = X1 + (size_t)pn * KPAD;
  const __bf16* src = pf + (size_t)p * DPAIR;
  for (int c = tid; c < KPAD; c += 256){
    float v;
    if (c < DPAIR)      v = lrelu(bf2f(src[c]) * rsA);
    else if (c < CTOT)  v = tnode[((b * NP + (c - DPAIR)) << 2) + n];
    else                v = 0.f;
    dst[c] = f2bf(v);
  }
}

// ---------------- big GEMM: Y[o, col] = sum_k W[o,k]*X[col,k], bf16 WMMA, f32 acc ----------------
// MODE 0 (SW): hg = g + leaky(Y + b), write bf16 hg[col][o]
// MODE 1 (CG): glbpost[col][o] = leaky(BN(Y + b))
// MODE 2 (DW): z = leaky(Y + b); atomicAdd(out2acc[col], sum_o ll_w[col&3][o]*z)
template<int MODE>
__global__ __launch_bounds__(256, 1)
void gemm_bf16_wmma(const float* __restrict__ W, const __bf16* __restrict__ X,
                    int M, int N, int K,
                    const float* __restrict__ bias,
                    const __bf16* __restrict__ pf, const float* __restrict__ vsam,
                    __bf16* __restrict__ hg,
                    const float* __restrict__ bng, const float* __restrict__ bnb,
                    const float* __restrict__ bnm, const float* __restrict__ bnv,
                    float* __restrict__ glbpost,
                    const float* __restrict__ llw, float* __restrict__ out2acc)
{
  __shared__ __align__(64) __bf16 lA[2][128 * 32];
  __shared__ __align__(64) __bf16 lB[2][128 * 32];
  const int tid  = threadIdx.x;
  const int lane = tid & 31;
  const int wave = tid >> 5;
  const int wm = (wave >> 1) * 32;   // 4 waves along M
  const int wn = (wave & 1) * 64;    // 2 waves along N
  const int m0 = blockIdx.y * 128;
  const int n0 = blockIdx.x * 128;

  v8f acc[2][4];
  #pragma unroll
  for (int i = 0; i < 2; ++i)
    #pragma unroll
    for (int j = 0; j < 4; ++j)
      #pragma unroll
      for (int v = 0; v < 8; ++v) acc[i][j][v] = 0.f;

  const int lrow = tid >> 1;           // 0..127
  const int lkb  = (tid & 1) * 16;     // 0 or 16

  auto stage = [&](int k0, int buf){
    // ---- W tile (fp32 -> bf16) ----
    const int grow = m0 + lrow;
    const float* ws = W + (size_t)grow * K + (k0 + lkb);
    if (grow < M && (k0 + lkb + 15) < K){
      const float4* w4 = (const float4*)ws;
      #pragma unroll
      for (int q = 0; q < 4; ++q){
        float4 f = w4[q];
        lA[buf][lrow * 32 + lkb + q * 4 + 0] = f2bf(f.x);
        lA[buf][lrow * 32 + lkb + q * 4 + 1] = f2bf(f.y);
        lA[buf][lrow * 32 + lkb + q * 4 + 2] = f2bf(f.z);
        lA[buf][lrow * 32 + lkb + q * 4 + 3] = f2bf(f.w);
      }
    } else {
      #pragma unroll
      for (int i = 0; i < 16; ++i){
        int k = k0 + lkb + i;
        float f = (grow < M && k < K) ? ws[i] : 0.f;
        lA[buf][lrow * 32 + lkb + i] = f2bf(f);
      }
    }
    // ---- X tile (bf16, Kpad-padded rows) ----
    const __bf16* xs = X + (size_t)(n0 + lrow) * KPAD + (k0 + lkb);
    uint4 u0 = ((const uint4*)xs)[0];
    uint4 u1 = ((const uint4*)xs)[1];
    *(uint4*)&lB[buf][lrow * 32 + lkb]     = u0;
    *(uint4*)&lB[buf][lrow * 32 + lkb + 8] = u1;
  };

  const int nK = (K + 31) / 32;  // 281
  stage(0, 0);
  __syncthreads();

  const int frow = lane & 15;
  const int fko  = (lane < 16) ? 0 : 16;

  for (int it = 0; it < nK; ++it){
    const int buf = it & 1;
    if (it + 1 < nK) stage((it + 1) * 32, buf ^ 1);
    if (it + 3 < nK){
      const int grow = m0 + lrow;
      if (grow < M) __builtin_prefetch(W + (size_t)grow * K + (it + 3) * 32 + lkb, 0, 1);
    }
    v16bf af[2], bfr[4];
    #pragma unroll
    for (int i = 0; i < 2; ++i)
      af[i] = *(const v16bf*)&lA[buf][(wm + i * 16 + frow) * 32 + fko];
    #pragma unroll
    for (int j = 0; j < 4; ++j)
      bfr[j] = *(const v16bf*)&lB[buf][(wn + j * 16 + frow) * 32 + fko];
    #pragma unroll
    for (int i = 0; i < 2; ++i)
      #pragma unroll
      for (int j = 0; j < 4; ++j)
        acc[i][j] = __builtin_amdgcn_wmma_f32_16x16x32_bf16(
            false, af[i], false, bfr[j], (short)0, acc[i][j], false, false);
    __syncthreads();
  }

  // ---- epilogue ----
  // C/D layout: lane<16 -> col = lane, rows = base + v; lane>=16 -> col = lane-16, rows = base+8+v
  const int rbase = wm + ((lane < 16) ? 0 : 8);
  #pragma unroll
  for (int j = 0; j < 4; ++j){
    const int col = n0 + wn + j * 16 + frow;
    if (MODE == 2){
      const int mloc = col & 3;
      float s = 0.f;
      #pragma unroll
      for (int i = 0; i < 2; ++i){
        const int rb = m0 + rbase + i * 16;
        #pragma unroll
        for (int v = 0; v < 8; ++v){
          const int o = rb + v;
          if (o < M){
            float z = lrelu(acc[i][j][v] + bias[o]);
            s += llw[(size_t)mloc * M + o] * z;
          }
        }
      }
      atomicAdd(out2acc + col, s);
    } else if (MODE == 0){
      const int p = col >> 2, n = col & 3, b = p >> 7;
      #pragma unroll
      for (int i = 0; i < 2; ++i){
        const int rb = m0 + rbase + i * 16;
        if (rb + 7 < M){
          __align__(16) __bf16 tmp[8];
          #pragma unroll
          for (int v = 0; v < 8; ++v){
            const int o = rb + v;
            float g = (o < DPAIR) ? bf2f(pf[(size_t)p * DPAIR + o])
                                  : vsam[((b * NP + (o - DPAIR)) << 2) + n];
            tmp[v] = f2bf(g + lrelu(acc[i][j][v] + bias[o]));
          }
          *(uint4*)&hg[(size_t)col * KPAD + rb] = *(const uint4*)tmp;
        } else {
          for (int v = 0; v < 8; ++v){
            const int o = rb + v;
            if (o < M){
              float g = (o < DPAIR) ? bf2f(pf[(size_t)p * DPAIR + o])
                                    : vsam[((b * NP + (o - DPAIR)) << 2) + n];
              hg[(size_t)col * KPAD + o] = f2bf(g + lrelu(acc[i][j][v] + bias[o]));
            }
          }
        }
      }
    } else { // MODE 1
      #pragma unroll
      for (int i = 0; i < 2; ++i){
        const int rb = m0 + rbase + i * 16;
        #pragma unroll
        for (int v = 0; v < 8; ++v){
          const int o = rb + v;
          if (o < M){
            float xv = acc[i][j][v] + bias[o];
            xv = (xv - bnm[o]) * rsqrtf(bnv[o] + 1e-5f) * bng[o] + bnb[o];
            glbpost[(size_t)col * M + o] = lrelu(xv);
          }
        }
      }
    }
  }
}

// ---------------- glbmean[p][k] = mean_n hg[p,k,n] (bf16) ----------------
__global__ __launch_bounds__(256) void k_glbmean(const __bf16* __restrict__ hg,
                                                 __bf16* __restrict__ gm)
{
  const int p = blockIdx.x, tid = threadIdx.x;
  for (int c = tid; c < KPAD; c += 256){
    float s = 0.f;
    if (c < CTOT){
      #pragma unroll
      for (int n = 0; n < NC; ++n) s += bf2f(hg[((size_t)(p * 4 + n)) * KPAD + c]);
      s *= 0.25f;
    }
    gm[(size_t)p * KPAD + c] = f2bf(s);
  }
}

// ---------------- adjA[p][o4] = sum_o co_w[o4, o] * glbpost[p, o] ----------------
__global__ __launch_bounds__(256) void k_adjA(const float* __restrict__ glbpost,
                                              const float* __restrict__ cow,
                                              float* __restrict__ adjA)
{
  __shared__ float red[NC * 256];
  const int p = blockIdx.x, tid = threadIdx.x;
  float a[NC] = {0.f, 0.f, 0.f, 0.f};
  const float* g = glbpost + (size_t)p * CTOT;
  for (int o = tid; o < CTOT; o += 256){
    float gv = g[o];
    #pragma unroll
    for (int odx = 0; odx < NC; ++odx) a[odx] += cow[(size_t)odx * 2 * CTOT + o] * gv;
  }
  #pragma unroll
  for (int odx = 0; odx < NC; ++odx) red[odx * 256 + tid] = a[odx];
  __syncthreads();
  for (int st = 128; st > 0; st >>= 1){
    if (tid < st){
      #pragma unroll
      for (int odx = 0; odx < NC; ++odx) red[odx * 256 + tid] += red[odx * 256 + tid + st];
    }
    __syncthreads();
  }
  if (tid < NC) adjA[p * 4 + tid] = red[tid * 256];
}

// ---------------- adjB[p][o4][n] = sum_c co_w[o4, C+c] * hg[p,c,n] ----------------
__global__ __launch_bounds__(256) void k_adjB(const __bf16* __restrict__ hg,
                                              const float* __restrict__ cow,
                                              float* __restrict__ adjB)
{
  __shared__ float red[NC * 256];
  const int pn = blockIdx.x;
  const int p = pn >> 2, n = pn & 3, tid = threadIdx.x;
  float a[NC] = {0.f, 0.f, 0.f, 0.f};
  const __bf16* h = hg + (size_t)pn * KPAD;
  for (int c = tid; c < CTOT; c += 256){
    float hv = bf2f(h[c]);
    #pragma unroll
    for (int odx = 0; odx < NC; ++odx) a[odx] += cow[(size_t)odx * 2 * CTOT + CTOT + c] * hv;
  }
  #pragma unroll
  for (int odx = 0; odx < NC; ++odx) red[odx * 256 + tid] = a[odx];
  __syncthreads();
  for (int st = 128; st > 0; st >>= 1){
    if (tid < st){
      #pragma unroll
      for (int odx = 0; odx < NC; ++odx) red[odx * 256 + tid] += red[odx * 256 + tid + st];
    }
    __syncthreads();
  }
  if (tid < NC) adjB[p * 16 + tid * 4 + n] = red[tid * 256];
}

// ---------------- hd[p,c,m] = leaky(sum_n hg[p,c,n]*sigmoid(adj)) ----------------
__global__ __launch_bounds__(256) void k_hd(const __bf16* __restrict__ hg,
                                            const float* __restrict__ adjA,
                                            const float* __restrict__ adjB,
                                            const float* __restrict__ cob,
                                            __bf16* __restrict__ hd)
{
  __shared__ float sm[16];
  const int p = blockIdx.x, tid = threadIdx.x;
  if (tid < 16){
    const int i = tid >> 2, jm = tid & 3;
    sm[tid] = sigmoidf_(adjA[p * 4 + i] + adjB[p * 16 + i * 4 + jm] + cob[i]);
  }
  __syncthreads();
  for (int c = tid; c < KPAD; c += 256){
    float hv[NC] = {0.f, 0.f, 0.f, 0.f};
    if (c < CTOT){
      #pragma unroll
      for (int n = 0; n < NC; ++n) hv[n] = bf2f(hg[((size_t)(p * 4 + n)) * KPAD + c]);
    }
    #pragma unroll
    for (int m = 0; m < NC; ++m){
      float v = 0.f;
      #pragma unroll
      for (int n2 = 0; n2 < NC; ++n2) v += hv[n2] * sm[n2 * 4 + m];
      hd[((size_t)(p * 4 + m)) * KPAD + c] = f2bf((c < CTOT) ? lrelu(v) : 0.f);
    }
  }
}

// ---------------- final: out = 0.5*(out1 + leaky(out2 + ll_b)) ----------------
__global__ __launch_bounds__(256) void k_final(const float* __restrict__ out1,
                                               const float* __restrict__ out2acc,
                                               const float* __restrict__ llb,
                                               float* __restrict__ out)
{
  const int i = blockIdx.x * 256 + threadIdx.x;   // 0..4095  == (b*128+pp)*4 + odx
  const int odx = i & 3;
  const int b = i >> 9;
  out[i] = 0.5f * (out1[b * 4 + odx] + lrelu(out2acc[i] + llb[odx]));
}

// ---------------- launcher ----------------
extern "C" void kernel_launch(void* const* d_in, const int* in_sizes, int n_in,
                              void* d_out, int out_size, void* d_ws, size_t ws_size,
                              hipStream_t stream) {
  (void)in_sizes; (void)n_in; (void)out_size; (void)ws_size;

  const float* x    = (const float*)d_in[0];
  const int*   rel  = (const int*)  d_in[1];
  const float* obj  = (const float*)d_in[2];
  const float* fcw  = (const float*)d_in[3];
  const float* ctw  = (const float*)d_in[4];
  const float* ctb  = (const float*)d_in[5];
  const float* A    = (const float*)d_in[6];
  const float* sww  = (const float*)d_in[7];
  const float* swb  = (const float*)d_in[8];
  const float* cgw  = (const float*)d_in[9];
  const float* cgb  = (const float*)d_in[10];
  const float* bng  = (const float*)d_in[11];
  const float* bnb  = (const float*)d_in[12];
  const float* bnm  = (const float*)d_in[13];
  const float* bnv  = (const float*)d_in[14];
  const float* cow  = (const float*)d_in[15];
  const float* cob  = (const float*)d_in[16];
  const float* dww  = (const float*)d_in[17];
  const float* dwb  = (const float*)d_in[18];
  const float* llw  = (const float*)d_in[19];
  const float* llb  = (const float*)d_in[20];
  float* out = (float*)d_out;

  // carve workspace
  size_t off = 0;
  auto carve = [&](size_t bytes) -> char* {
    char* p = (char*)d_ws + off;
    off += (bytes + 255) & ~(size_t)255;
    return p;
  };
  float*  yws     = (float*) carve((size_t)NB * NC * HW * 4);
  float*  out1ws  = (float*) carve((size_t)NB * NC * 4);
  float*  vws     = (float*) carve((size_t)NB * NP * NC * 4);
  float*  tnode   = (float*) carve((size_t)NB * NP * NC * 4);
  __bf16* pfws    = (__bf16*)carve((size_t)BP * DPAIR * 2);
  __bf16* X1ws    = (__bf16*)carve((size_t)NCOL * KPAD * 2);   // later reused as hd
  __bf16* hgws    = (__bf16*)carve((size_t)NCOL * KPAD * 2);
  __bf16* gmws    = (__bf16*)carve((size_t)BP * KPAD * 2);
  float*  glbpost = (float*) carve((size_t)BP * CTOT * 4);
  float*  adjA    = (float*) carve((size_t)BP * NC * 4);
  float*  adjB    = (float*) carve((size_t)BP * NC * NC * 4);
  float*  out2acc = (float*) carve((size_t)NCOL * 4);
  __bf16* hdws    = X1ws;  // alias: X1 is dead after the SW GEMM

  hipMemsetAsync(out2acc, 0, (size_t)NCOL * 4, stream);

  k_y_out1<<<NB, 256, 0, stream>>>(x, fcw, yws, out1ws);
  k_vsam  <<<NB * NP, 256, 0, stream>>>(x, ctw, ctb, yws, A, vws, tnode);
  k_gather<<<BP, 256, 0, stream>>>(rel, obj, pfws);
  k_buildX1<<<NCOL, 256, 0, stream>>>(pfws, tnode, A, X1ws);

  // SW GEMM: hg = g + leaky(sw_w @ X1 + sw_b)
  gemm_bf16_wmma<0><<<dim3(NCOL / 128, (CTOT + 127) / 128), 256, 0, stream>>>(
      sww, X1ws, CTOT, NCOL, CTOT, swb,
      pfws, vws, hgws,
      nullptr, nullptr, nullptr, nullptr, nullptr,
      nullptr, nullptr);

  k_glbmean<<<BP, 256, 0, stream>>>(hgws, gmws);

  // CG GEMM: glbpost = leaky(BN(cg_w @ glbmean + cg_b))
  gemm_bf16_wmma<1><<<dim3(BP / 128, (CTOT + 127) / 128), 256, 0, stream>>>(
      cgw, gmws, CTOT, BP, CTOT, cgb,
      nullptr, nullptr, nullptr,
      bng, bnb, bnm, bnv, glbpost,
      nullptr, nullptr);

  k_adjA<<<BP, 256, 0, stream>>>(glbpost, cow, adjA);
  k_adjB<<<NCOL, 256, 0, stream>>>(hgws, cow, adjB);
  k_hd  <<<BP, 256, 0, stream>>>(hgws, adjA, adjB, cob, hdws);

  // DW GEMM: z = leaky(dw_w @ hd + dw_b), fused ll_w diagonal reduction
  gemm_bf16_wmma<2><<<dim3(NCOL / 128, (CTOT + 127) / 128), 256, 0, stream>>>(
      dww, hdws, CTOT, NCOL, CTOT, dwb,
      nullptr, nullptr, nullptr,
      nullptr, nullptr, nullptr, nullptr, nullptr,
      llw, out2acc);

  k_final<<<NCOL / 256, 256, 0, stream>>>(out1ws, out2acc, llb, out);
}
